// ourModel_basis_18983755448416
// MI455X (gfx1250) — matrix operand order repaired and checked by reading, not compiled, not run on hardware
//
#include <hip/hip_runtime.h>

// MI455X / gfx1250 implementation.
// einsum('nf,kef->nke') is cast as GEMM (50000x64) @ (64x1024) through
// v_wmma_f32_16x16x32_f16 (f16 in, f32 accumulate). Persistent GEMM blocks keep the
// whole B operand in VGPRs (preloaded once via LDS), so the steady-state loop is
// A-loads + 8 WMMAs + stores. T is materialized f16 (halves gather bandwidth);
// the edge scatter uses f32 global atomics into a 12.8MB L2-resident accumulator.

#define NNODES 50000
#define NEDGES 800000
#define DFEAT  64
#define INFEAT 128
#define TCOLS  1024   // 16 basis matrices * 64 features
#define EPSV   1e-5f

typedef __attribute__((ext_vector_type(16))) _Float16 v16h;
typedef __attribute__((ext_vector_type(8)))  _Float16 v8h;
typedef __attribute__((ext_vector_type(2)))  _Float16 v2h;
typedef __attribute__((ext_vector_type(8)))  float    v8f;

__global__ void k_f32_to_f16(const float* __restrict__ s, _Float16* __restrict__ d, int n) {
    int i = blockIdx.x * blockDim.x + threadIdx.x;
    if (i < n) d[i] = (_Float16)s[i];
}

// Wt[n*kdim + k] = W[k*DFEAT + n]  (B stored column-major for per-lane contiguous K)
__global__ void k_pack_w(const float* __restrict__ W, _Float16* __restrict__ Wt, int kdim) {
    int i = blockIdx.x * blockDim.x + threadIdx.x;
    if (i < kdim * DFEAT) {
        int n = i / kdim, k = i - n * kdim;
        Wt[i] = (_Float16)W[k * DFEAT + n];
    }
}

__global__ void k_zero(float* __restrict__ p, int n) {
    int i = blockIdx.x * blockDim.x + threadIdx.x;
    if (i < n) p[i] = 0.0f;
}

// C[nrows x ncols] = A[nrows x KDIM] @ B, B given column-major as Bt[ncols x KDIM].
// Block = 256 threads = 8 waves; wave tile = 16 rows x 64 cols; block = 128 rows/iter.
// Persistent: B tile staged to LDS once, wave's B fragments preloaded to VGPRs once,
// then a grid-stride loop over row tiles. nrows must be a multiple of 16 (50000 is).
template<int KDIM, bool LINEAR>
__launch_bounds__(256)
__global__ void k_gemm_wmma(const _Float16* __restrict__ A,
                            const _Float16* __restrict__ Bt,
                            int nrows,
                            const float* __restrict__ bias,   // used when LINEAR
                            float*    __restrict__ outF32,    // used when LINEAR
                            _Float16* __restrict__ outF16,
                            int ldo) {
    constexpr int KSTEPS = KDIM / 32;
    __shared__ __align__(32) _Float16 ldsB[64 * KDIM];
    const int tid  = threadIdx.x;
    const int lane = tid & 31;
    const int wave = tid >> 5;
    const int colBase = blockIdx.y * 64;

    // Stage B tile (64 cols x KDIM halves) into LDS once.
    constexpr int CHUNKS = (64 * KDIM) >> 3;
    #pragma unroll
    for (int c = 0; c < CHUNKS / 256 + 1; ++c) {
        int idx = c * 256 + tid;
        if (idx < CHUNKS) {
            int off = idx << 3;
            *(v8h*)(&ldsB[off]) = *(const v8h*)(&Bt[(size_t)colBase * KDIM + off]);
        }
    }
    __syncthreads();

    // B fragment layout: lane&15 = column N; lanes<16 hold K=kb..kb+15,
    // lanes>=16 hold K=kb+16..kb+31 (contiguous 16 halves in column-major Bt).
    const int bcol = lane & 15;
    const int bK   = (lane >> 4) * 16;
    v16h bfrag[4 * KSTEPS];
    #pragma unroll
    for (int nt = 0; nt < 4; ++nt)
        #pragma unroll
        for (int ks = 0; ks < KSTEPS; ++ks)
            bfrag[nt * KSTEPS + ks] =
                *(const v16h*)(&ldsB[(nt * 16 + bcol) * KDIM + ks * 32 + bK]);

    // A fragment layout: lanes 0-15: row M=lane, K=kb..kb+15;
    // lanes 16-31: row M=lane-16, K=kb+16..kb+31.
    const int aK0  = (lane >> 4) * 16;
    const int rOff = (lane >> 4) * 8;

    for (int rowBase = blockIdx.x * 128 + wave * 16;
         rowBase + 16 <= nrows;
         rowBase += gridDim.x * 128) {
        const _Float16* aptr = A + (size_t)(rowBase + (lane & 15)) * KDIM;

        v16h afrag[KSTEPS];
        #pragma unroll
        for (int ks = 0; ks < KSTEPS; ++ks) {
            v8h a0 = *(const v8h*)(aptr + ks * 32 + aK0);
            v8h a1 = *(const v8h*)(aptr + ks * 32 + 8 + aK0);
            #pragma unroll
            for (int i = 0; i < 8; ++i) { afrag[ks][i] = a0[i]; afrag[ks][8 + i] = a1[i]; }
        }

        v8f acc[4] = {};
        #pragma unroll
        for (int ks = 0; ks < KSTEPS; ++ks)
            #pragma unroll
            for (int nt = 0; nt < 4; ++nt)
                acc[nt] = __builtin_amdgcn_wmma_f32_16x16x32_f16(
                    false, afrag[ks], false, bfrag[nt * KSTEPS + ks],
                    (short)0, acc[nt], false, false);

        // D layout: lanes 0-15 VGPR r -> row M=r; lanes 16-31 -> row M=r+8; col = lane&15.
        #pragma unroll
        for (int nt = 0; nt < 4; ++nt) {
            const int col = colBase + nt * 16 + (lane & 15);
            float bv = 0.0f;
            if (LINEAR) bv = bias[col];
            #pragma unroll
            for (int r = 0; r < 8; ++r) {
                const size_t idx = (size_t)(rowBase + rOff + r) * ldo + col;
                const float v = acc[nt][r] + bv;
                if (LINEAR) outF32[idx] = v;
                outF16[idx] = (_Float16)v;
            }
        }
    }
}

// 32 threads per edge; each thread handles 2 features (one b32 load of 2 halves + 2 f32 atomics).
__global__ void k_scatter(const _Float16* __restrict__ T,
                          const int* __restrict__ src, const int* __restrict__ dst,
                          const int* __restrict__ labels, const int* __restrict__ bidx,
                          float* __restrict__ agg) {
    int gid = blockIdx.x * blockDim.x + threadIdx.x;
    int e = gid >> 5;
    if (e >= NEDGES) return;
    int fp = (gid & 31) * 2;
    int s = src[e], d = dst[e];
    int combo = (labels[d] * 2 + labels[s]) * 4 + bidx[e];
    v2h m = *(const v2h*)(T + (size_t)s * TCOLS + combo * DFEAT + fp);
    float* ap = agg + (size_t)d * DFEAT + fp;
    atomicAdd(ap,     (float)m[0]);
    atomicAdd(ap + 1, (float)m[1]);
}

// Per-feature sum / sum-of-squares of (H + agg) over all nodes.
__launch_bounds__(256)
__global__ void k_stats(const float* __restrict__ H, const float* __restrict__ agg,
                        float* __restrict__ stats) {
    __shared__ float ls[256], ls2[256];
    const int tid = threadIdx.x;
    const int f = tid & 63, rg = tid >> 6;
    float s = 0.0f, s2 = 0.0f;
    for (int row = blockIdx.x * 4 + rg; row < NNODES; row += gridDim.x * 4) {
        float v = H[(size_t)row * DFEAT + f] + agg[(size_t)row * DFEAT + f];
        s += v; s2 += v * v;
    }
    ls[tid] = s; ls2[tid] = s2;
    __syncthreads();
    if (tid < 64) {
        s  = ls[tid]  + ls[tid + 64]  + ls[tid + 128]  + ls[tid + 192];
        s2 = ls2[tid] + ls2[tid + 64] + ls2[tid + 128] + ls2[tid + 192];
        atomicAdd(&stats[f], s);
        atomicAdd(&stats[64 + f], s2);
    }
}

__global__ void k_norm(const float* __restrict__ H, const float* __restrict__ agg,
                       const float* __restrict__ stats,
                       const float* __restrict__ g, const float* __restrict__ be,
                       float* __restrict__ act, _Float16* __restrict__ a16) {
    int i = blockIdx.x * blockDim.x + threadIdx.x;
    if (i >= NNODES * DFEAT) return;
    int f = i & 63;
    float v = H[i] + agg[i];
    float mu = stats[f] * (1.0f / NNODES);
    float var = stats[64 + f] * (1.0f / NNODES) - mu * mu;
    float y = (v - mu) * rsqrtf(var + EPSV) * g[f] + be[f];
    y = fmaxf(y, 0.0f);
    act[i] = y;
    a16[i] = (_Float16)y;
}

__global__ void k_head(const float* __restrict__ act, const float* __restrict__ resW,
                       const float* __restrict__ resb, float* __restrict__ out) {
    int n = blockIdx.x * blockDim.x + threadIdx.x;
    if (n >= NNODES) return;
    float a0 = resb[0], a1 = resb[1];
    const float* h = act + (size_t)n * DFEAT;
    #pragma unroll 8
    for (int f = 0; f < DFEAT; ++f) {
        float v = h[f];
        a0 += v * resW[f * 2];
        a1 += v * resW[f * 2 + 1];
    }
    out[n * 2]     = a0;
    out[n * 2 + 1] = a1;
}

extern "C" void kernel_launch(void* const* d_in, const int* in_sizes, int n_in,
                              void* d_out, int out_size, void* d_ws, size_t ws_size,
                              hipStream_t stream) {
    (void)in_sizes; (void)n_in; (void)out_size; (void)ws_size;
    const float* x      = (const float*)d_in[0];
    const float* cross  = (const float*)d_in[1];
    const int*   src    = (const int*)d_in[2];
    const int*   dst    = (const int*)d_in[3];
    const int*   labels = (const int*)d_in[4];
    const int*   bidx   = (const int*)d_in[5];
    const float* W[3]  = {(const float*)d_in[6],  (const float*)d_in[10], (const float*)d_in[14]};
    const float* bb[3] = {(const float*)d_in[7],  (const float*)d_in[11], (const float*)d_in[15]};
    const float* gg[3] = {(const float*)d_in[8],  (const float*)d_in[12], (const float*)d_in[16]};
    const float* be[3] = {(const float*)d_in[9],  (const float*)d_in[13], (const float*)d_in[17]};
    const float* resW  = (const float*)d_in[18];
    const float* resb  = (const float*)d_in[19];
    float* out = (float*)d_out;

    char* ws = (char*)d_ws;
    size_t off = 0;
    auto alloc = [&](size_t bytes) -> void* {
        size_t a = (off + 255) & ~(size_t)255;
        off = a + bytes;
        return (void*)(ws + a);
    };
    _Float16* A16   = (_Float16*)alloc((size_t)NNODES * INFEAT * 2);
    _Float16* H16   = (_Float16*)alloc((size_t)NNODES * DFEAT * 2);
    _Float16* Mt16  = (_Float16*)alloc((size_t)TCOLS * DFEAT * 2);
    _Float16* Wt16  = (_Float16*)alloc((size_t)INFEAT * DFEAT * 2);
    float*    H     = (float*)alloc((size_t)NNODES * DFEAT * 4);
    float*    AGG   = (float*)alloc((size_t)NNODES * DFEAT * 4);
    float*    ACT   = (float*)alloc((size_t)NNODES * DFEAT * 4);
    float*    stats = (float*)alloc(128 * 4);
    _Float16* T16   = (_Float16*)alloc((size_t)NNODES * TCOLS * 2);

    // f16 conversions: activations (x) and the flattened 64x1024 basis matrix.
    // cross_basis flat (k,e,f) == Bt[col=k*64+e][f] column-major layout directly.
    k_f32_to_f16<<<(NNODES * INFEAT + 255) / 256, 256, 0, stream>>>(x, A16, NNODES * INFEAT);
    k_f32_to_f16<<<(TCOLS * DFEAT + 255) / 256, 256, 0, stream>>>(cross, Mt16, TCOLS * DFEAT);

    for (int l = 0; l < 3; ++l) {
        const int kdim = (l == 0) ? INFEAT : DFEAT;
        k_pack_w<<<(kdim * DFEAT + 255) / 256, 256, 0, stream>>>(W[l], Wt16, kdim);

        dim3 gl(128, 1);
        if (l == 0)
            k_gemm_wmma<INFEAT, true><<<gl, 256, 0, stream>>>(A16, Wt16, NNODES, bb[l], H, H16, DFEAT);
        else
            k_gemm_wmma<DFEAT, true><<<gl, 256, 0, stream>>>(A16, Wt16, NNODES, bb[l], H, H16, DFEAT);

        dim3 gt(128, TCOLS / 64);
        k_gemm_wmma<DFEAT, false><<<gt, 256, 0, stream>>>(H16, Mt16, NNODES, nullptr, nullptr, T16, TCOLS);

        k_zero<<<(NNODES * DFEAT + 255) / 256, 256, 0, stream>>>(AGG, NNODES * DFEAT);
        k_zero<<<1, 128, 0, stream>>>(stats, 128);

        k_scatter<<<(NEDGES * 32 + 255) / 256, 256, 0, stream>>>(T16, src, dst, labels, bidx, AGG);
        k_stats<<<240, 256, 0, stream>>>(H, AGG, stats);
        k_norm<<<(NNODES * DFEAT + 255) / 256, 256, 0, stream>>>(H, AGG, stats, gg[l], be[l], ACT, A16);
    }
    k_head<<<(NNODES + 255) / 256, 256, 0, stream>>>(ACT, resW, resb, out);
}
